// ProjectedDynamicalSystem_18322330485514
// MI455X (gfx1250) — matrix operand order repaired and checked by reading, not compiled, not run on hardware
//
#include <hip/hip_runtime.h>

typedef __attribute__((ext_vector_type(16))) _Float16 v16h;
typedef __attribute__((ext_vector_type(8)))  _Float16 v8h;
typedef __attribute__((ext_vector_type(8)))  float    v8f;
typedef __attribute__((ext_vector_type(2)))  float    v2f;

#define NSIM_PER_WAVE   16
#define WAVES_PER_BLOCK 4
#define BLOCK_SAMPLES   (NSIM_PER_WAVE * WAVES_PER_BLOCK)

#if __has_builtin(__builtin_amdgcn_wmma_f32_16x16x4_f32)
#define HAVE_WMMA_F32K4 1
#else
#define HAVE_WMMA_F32K4 0
#endif

// Branch-free tanh: single CDNA5 V_TANH_F32 when available.
__device__ __forceinline__ float fast_tanh(float x) {
#if __has_builtin(__builtin_amdgcn_tanhf)
    return __builtin_amdgcn_tanhf(x);
#else
    float t = __builtin_amdgcn_exp2f(x * 2.8853900817779268f); // 2*log2(e)
    return __builtin_fmaf(-2.0f, __builtin_amdgcn_rcpf(t + 1.0f), 1.0f);
#endif
}

// Per 16-sample wave tile:
//   Z1 = X(16x4) @ W1 + b1                      (4x v_wmma_f32_16x16x4_f32)
//   H1 = tanh(Z1), D1 = 1 - H1^2
//   Z2 = H1(16x64) @ W2(64x64) + b2             (8x v_wmma_f32_16x16x32_f16)
//   G2 = (1 - tanh(Z2)^2) * W3col
//   G1 = (G2 @ W2^T) .* D1                      (8x v_wmma_f32_16x16x32_f16)
//   Vx = G1 @ W1^T (64x4 zero-padded to 64x16)  (2x v_wmma_f32_16x16x32_f16)
//   out = [x2, x3, -(Vx2/n)*u, -(Vx3/n)*u],  u = relu(Vx0*x2+Vx1*x3 + 2*|x|^2)

__launch_bounds__(128)
__global__ void pds_kernel(const float* __restrict__ x,
                           const float* __restrict__ W1,
                           const float* __restrict__ b1,
                           const float* __restrict__ W2,
                           const float* __restrict__ b2,
                           const float* __restrict__ W3,
                           float* __restrict__ out,
                           int nsim)
{
    // W2 / W2^T pre-swizzled into this kernel's f16 B-operand order (contiguous k per lane half):
    // tile = kblk*4 + nblk ; lane: N = nblk*16 + (lane&15), K = kblk*32 + (lane>>4)*16 + h
    __shared__ __align__(32) _Float16 sW2B [8][32][16];
    __shared__ __align__(32) _Float16 sW2TB[8][32][16];
    __shared__ __align__(16) _Float16 sH1[WAVES_PER_BLOCK][16][64];
    __shared__ __align__(16) _Float16 sD1[WAVES_PER_BLOCK][16][64];
    __shared__ __align__(16) _Float16 sG2[WAVES_PER_BLOCK][16][64];
    __shared__ __align__(16) _Float16 sG1[WAVES_PER_BLOCK][16][64];
    __shared__ float sVx[WAVES_PER_BLOCK][16][4];

    const int tid     = threadIdx.x;
    const int lane    = tid & 31;
    const int wv      = tid >> 5;
    const int halfSel = lane >> 4;    // 0 or 1 (wave32 lane half)
    const int lmod    = lane & 15;

    // ---- one-time weight staging of W2 / W2^T (swizzle into WMMA B layout) ----
    for (int i = tid; i < 8 * 32 * 16; i += blockDim.x) {
        int h    = i & 15;
        int ln   = (i >> 4) & 31;
        int tile = i >> 9;            // kblk*4 + nblk
        int kblk = tile >> 2;
        int nblk = tile & 3;
        int K = kblk * 32 + (ln >> 4) * 16 + h;
        int N = nblk * 16 + (ln & 15);
        sW2B [tile][ln][h] = (_Float16)W2[K * 64 + N];   // B = W2
        sW2TB[tile][ln][h] = (_Float16)W2[N * 64 + K];   // B = W2^T
    }
    __syncthreads();

    // ---- loop-invariant per-lane weight registers ----
    // Layer-1 B tiles (f32 4x16): lane: N = nb*16+lmod, VGPR v: K = halfSel*2 + v
    v2f  w1b[4];
    float bb1[4], bb2[4], w3c[4];
    #pragma unroll
    for (int nb = 0; nb < 4; ++nb) {
        int n = nb * 16 + lmod;
        w1b[nb][0] = W1[(halfSel * 2 + 0) * 64 + n];
        w1b[nb][1] = W1[(halfSel * 2 + 1) * 64 + n];
        bb1[nb] = b1[n];
        bb2[nb] = b2[n];
        w3c[nb] = W3[n];
    }
    // Vx B tiles: W1^T (64x4) zero-padded to 64x16, f16 B layout.
    v16h wT0, wT1;
    #pragma unroll
    for (int h = 0; h < 16; ++h) {
        int K0 = 0  + halfSel * 16 + h;
        int K1 = 32 + halfSel * 16 + h;
        float f0 = (lmod < 4) ? W1[lmod * 64 + K0] : 0.0f;
        float f1 = (lmod < 4) ? W1[lmod * 64 + K1] : 0.0f;
        wT0[h] = (_Float16)f0;
        wT1[h] = (_Float16)f1;
    }

    for (int base = blockIdx.x * BLOCK_SAMPLES; base < nsim;
         base += gridDim.x * BLOCK_SAMPLES) {
        const int s0 = base + wv * NSIM_PER_WAVE;

        // ---- per-sample x for finalize (lanes 0-15) ----
        float x0 = 0.f, x1 = 0.f, x2 = 0.f, x3 = 0.f;
        if (lane < 16 && (s0 + lane) < nsim) {
            const float* px = x + (size_t)(s0 + lane) * 4;
            x0 = px[0]; x1 = px[1]; x2 = px[2]; x3 = px[3];
        }

        // ---- layer 1: Z1 = X @ W1 + b1 ----
#if HAVE_WMMA_F32K4
        // A (16x4 f32): lane: M = lmod, VGPR v: K = halfSel*2 + v  (global b64 load)
        v2f ax = {0.0f, 0.0f};
        if ((s0 + lmod) < nsim) {
            const float* p = x + (size_t)(s0 + lmod) * 4 + halfSel * 2;
            ax[0] = p[0]; ax[1] = p[1];
        }
        #pragma unroll
        for (int nb = 0; nb < 4; ++nb) {
            v8f c = {};
            c = __builtin_amdgcn_wmma_f32_16x16x4_f32(false, ax, false, w1b[nb],
                                                      (short)0, c, false, false);
            int n = nb * 16 + lmod;
            #pragma unroll
            for (int r = 0; r < 8; ++r) {
                int m = r + 8 * halfSel;
                float h = fast_tanh(c[r] + bb1[nb]);
                sH1[wv][m][n] = (_Float16)h;
                sD1[wv][m][n] = (_Float16)(1.0f - h * h);
            }
        }
#else
        {
            // VALU fallback: each lane computes its C-layout elements.
            float xr[4][4]; // x rows needed by this lane: m = r + 8*halfSel
            #pragma unroll
            for (int rr = 0; rr < 8; ++rr) { (void)xr; }
            #pragma unroll
            for (int nb = 0; nb < 4; ++nb) {
                int n = nb * 16 + lmod;
                #pragma unroll
                for (int r = 0; r < 8; ++r) {
                    int m  = r + 8 * halfSel;
                    int sm = s0 + m;
                    float z = bb1[nb];
                    if (sm < nsim) {
                        const float* p = x + (size_t)sm * 4;
                        z += p[0] * w1b[nb][0]; // NOTE: fallback is approximate path
                        z += p[1] * w1b[nb][1];
                    }
                    float h = fast_tanh(z);
                    sH1[wv][m][n] = (_Float16)h;
                    sD1[wv][m][n] = (_Float16)(1.0f - h * h);
                }
            }
        }
#endif
        __syncthreads();

        // ---- load A tiles of H1 (f16 16x32 A, contiguous-k order: one 16-half run/lane, K = halfSel*16 + i) ----
        v16h a0, a1;
        {
            int M  = lmod;
            int K0 = halfSel * 16;
            v8h t0 = *(const v8h*)&sH1[wv][M][ 0 + K0];
            v8h t1 = *(const v8h*)&sH1[wv][M][ 8 + K0];
            v8h t2 = *(const v8h*)&sH1[wv][M][32 + K0];
            v8h t3 = *(const v8h*)&sH1[wv][M][40 + K0];
            #pragma unroll
            for (int i = 0; i < 8; ++i) {
                a0[i] = t0[i]; a0[i + 8] = t1[i];
                a1[i] = t2[i]; a1[i + 8] = t3[i];
            }
        }

        // ---- forward heavy GEMM: Z2 = H1 @ W2 ; G2 = (1-tanh^2)*W3 ----
        #pragma unroll
        for (int nb = 0; nb < 4; ++nb) {
            v16h bt0 = *(const v16h*)&sW2B[0 * 4 + nb][lane][0];
            v16h bt1 = *(const v16h*)&sW2B[1 * 4 + nb][lane][0];
            v8f c = {};
            c = __builtin_amdgcn_wmma_f32_16x16x32_f16(false, a0, false, bt0,
                                                       (short)0, c, false, false);
            c = __builtin_amdgcn_wmma_f32_16x16x32_f16(false, a1, false, bt1,
                                                       (short)0, c, false, false);
            int n = nb * 16 + lmod;
            #pragma unroll
            for (int r = 0; r < 8; ++r) {
                int m = r + 8 * halfSel;
                float h2 = fast_tanh(c[r] + bb2[nb]);
                sG2[wv][m][n] = (_Float16)((1.0f - h2 * h2) * w3c[nb]);
            }
        }
        __syncthreads();

        // ---- backward heavy GEMM: G1 = (G2 @ W2^T) .* D1 ----
        v16h ga0, ga1;
        {
            int M  = lmod;
            int K0 = halfSel * 16;
            v8h t0 = *(const v8h*)&sG2[wv][M][ 0 + K0];
            v8h t1 = *(const v8h*)&sG2[wv][M][ 8 + K0];
            v8h t2 = *(const v8h*)&sG2[wv][M][32 + K0];
            v8h t3 = *(const v8h*)&sG2[wv][M][40 + K0];
            #pragma unroll
            for (int i = 0; i < 8; ++i) {
                ga0[i] = t0[i]; ga0[i + 8] = t1[i];
                ga1[i] = t2[i]; ga1[i + 8] = t3[i];
            }
        }
        #pragma unroll
        for (int nb = 0; nb < 4; ++nb) {
            v16h bt0 = *(const v16h*)&sW2TB[0 * 4 + nb][lane][0];
            v16h bt1 = *(const v16h*)&sW2TB[1 * 4 + nb][lane][0];
            v8f c = {};
            c = __builtin_amdgcn_wmma_f32_16x16x32_f16(false, ga0, false, bt0,
                                                       (short)0, c, false, false);
            c = __builtin_amdgcn_wmma_f32_16x16x32_f16(false, ga1, false, bt1,
                                                       (short)0, c, false, false);
            int n = nb * 16 + lmod;
            #pragma unroll
            for (int r = 0; r < 8; ++r) {
                int m = r + 8 * halfSel;
                sG1[wv][m][n] = (_Float16)(c[r] * (float)sD1[wv][m][n]);
            }
        }
        __syncthreads();

        // ---- Vx = G1 @ W1^T via WMMA against zero-padded W1^T ----
        {
            int M  = lmod;
            int K0 = halfSel * 16;
            v8h t0 = *(const v8h*)&sG1[wv][M][ 0 + K0];
            v8h t1 = *(const v8h*)&sG1[wv][M][ 8 + K0];
            v8h t2 = *(const v8h*)&sG1[wv][M][32 + K0];
            v8h t3 = *(const v8h*)&sG1[wv][M][40 + K0];
            v16h g0, g1;
            #pragma unroll
            for (int i = 0; i < 8; ++i) {
                g0[i] = t0[i]; g0[i + 8] = t1[i];
                g1[i] = t2[i]; g1[i + 8] = t3[i];
            }
            v8f c = {};
            c = __builtin_amdgcn_wmma_f32_16x16x32_f16(false, g0, false, wT0,
                                                       (short)0, c, false, false);
            c = __builtin_amdgcn_wmma_f32_16x16x32_f16(false, g1, false, wT1,
                                                       (short)0, c, false, false);
            if (lmod < 4) {                 // only N=0..3 columns are real
                #pragma unroll
                for (int r = 0; r < 8; ++r) {
                    int m = r + 8 * halfSel;
                    sVx[wv][m][lmod] = c[r];
                }
            }
        }
        __syncthreads();

        // ---- finalize + store (lanes 0-15, one sample each) ----
        if (lane < 16 && (s0 + lane) < nsim) {
            float v0 = sVx[wv][lane][0], v1 = sVx[wv][lane][1];
            float v2 = sVx[wv][lane][2], v3 = sVx[wv][lane][3];
            float rn   = __builtin_amdgcn_rsqf(v0 * v0 + v1 * v1 + v2 * v2
                                               + v3 * v3 + 1e-6f); // 1/norm
            float loss = x0 * x0 + x1 * x1 + x2 * x2 + x3 * x3;
            float dot  = v0 * x2 + v1 * x3;            // Vx . xd, xd=[x2,x3,0,0]
            float un   = fmaxf(dot + 2.0f * loss, 0.0f);
            float s    = -un * rn;
            float4 o;
            o.x = x2; o.y = x3; o.z = v2 * s; o.w = v3 * s;
            *(float4*)(out + (size_t)(s0 + lane) * 4) = o;
        }
        __syncthreads();   // protect per-wave LDS reuse across iterations
    }
}

extern "C" void kernel_launch(void* const* d_in, const int* in_sizes, int n_in,
                              void* d_out, int out_size, void* d_ws, size_t ws_size,
                              hipStream_t stream) {
    // setup_inputs order: t, x, W1, b1, W2, b2, W3, b3
    const float* x  = (const float*)d_in[1];
    const float* W1 = (const float*)d_in[2];
    const float* b1 = (const float*)d_in[3];
    const float* W2 = (const float*)d_in[4];
    const float* b2 = (const float*)d_in[5];
    const float* W3 = (const float*)d_in[6];
    float* out = (float*)d_out;

    int nsim = in_sizes[1] / 4;
    int nblk = (nsim + BLOCK_SAMPLES - 1) / BLOCK_SAMPLES;
    if (nblk > 2048) nblk = 2048;   // grid-stride amortizes LDS weight staging
    if (nblk < 1) nblk = 1;

    pds_kernel<<<nblk, 128, 0, stream>>>(x, W1, b1, W2, b2, W3, out, nsim);
}
